// walkGenerateNet_64871186039438
// MI455X (gfx1250) — matrix operand order repaired and checked by reading, not compiled, not run on hardware
//
#include <hip/hip_runtime.h>
#include <hip/hip_bf16.h>

typedef __bf16 bf16_t;
typedef __attribute__((ext_vector_type(16))) __bf16 v16bf;
typedef __attribute__((ext_vector_type(8)))  __bf16 v8bf;
typedef __attribute__((ext_vector_type(8)))  float   v8f;

#define B_ROWS 512
#define T_STEPS 85
#define D_IN 27

__device__ __forceinline__ v16bf load_afrag(const bf16_t* __restrict__ ar,
                                            int k0, int half) {
  // A row-major: row m, K = k0 + 16*half .. +15 contiguous (matches B)
  return *(const v16bf*)(ar + k0 + 16 * half);
}

__device__ __forceinline__ v16bf load_bfrag(const bf16_t* __restrict__ br,
                                            int k0, int half) {
  // WT row-major [N][Ks]: col n, K = k0 + 16*half .. +15 contiguous
  return *(const v16bf*)(br + k0 + 16 * half);
}

__device__ __forceinline__ v8f wmma_bf16(v16bf a, v16bf b, v8f c) {
  return __builtin_amdgcn_wmma_f32_16x16x32_bf16(false, a, false, b,
                                                 (short)0, c, false, false);
}

// ---------------------------------------------------------------------------
// bf16 WMMA GEMM, 2x2 tile register blocking: each wave -> 32x32 of C.
//   C[512 x N] = act(A[512 x Ks] @ WT^T + bias)
//   A  : row-major [512][Ks], Ks multiple of 32 (zero padded)
//   WT : W transposed, row-major [Npad16][Ks] (zero padded)
// MODE 0: relu, store bf16 [row][outStride]
// MODE 1: no act, store bf16 [row][outStride]
// MODE 2: relu, store 3 ew-scaled bf16 copies [row][3N] (fused MoE input)
// MODE 3: final (N=27): f32 -> d_out[:,t,:], lastF, bf16 gateA (pad 0)
// ---------------------------------------------------------------------------
template<int MODE>
__device__ __forceinline__ void epilogue_tile(
    const v8f& c, int rowBase, int colBase, int half, int l15,
    const float* __restrict__ bias, int N,
    bf16_t* __restrict__ outB, int outStride,
    const float* __restrict__ ew, bf16_t* __restrict__ outT,
    float* __restrict__ outF, float* __restrict__ lastF,
    bf16_t* __restrict__ gateA, int t)
{
  const int col = colBase + l15;
  const float bv = (col < N) ? bias[col] : 0.f;
#pragma unroll
  for (int j = 0; j < 8; ++j) {
    const int row = rowBase + j + 8 * half;   // ISA C/D layout
    float v = c[j] + bv;
    if (MODE == 0) {
      v = fmaxf(v, 0.f);
      outB[(size_t)row * outStride + col] = (bf16_t)v;
    } else if (MODE == 1) {
      outB[(size_t)row * outStride + col] = (bf16_t)v;
    } else if (MODE == 2) {
      v = fmaxf(v, 0.f);
      const size_t base = (size_t)row * (3 * N);
      outT[base +         col] = (bf16_t)(ew[row * 3 + 0] * v);
      outT[base +   N   + col] = (bf16_t)(ew[row * 3 + 1] * v);
      outT[base + 2 * N + col] = (bf16_t)(ew[row * 3 + 2] * v);
    } else { // MODE 3 : N = 27
      if (col < D_IN) {
        outF[((size_t)row * T_STEPS + t) * D_IN + col] = v;
        lastF[row * D_IN + col] = v;
        gateA[row * 32 + col] = (bf16_t)v;
      } else if (col < 32) {
        gateA[row * 32 + col] = (bf16_t)0.f;  // keep gate-A K padding zero
      }
    }
  }
}

template<int MODE>
__global__ __launch_bounds__(256) void gemm_bf16_wmma(
    const bf16_t* __restrict__ A, int Ks,
    const bf16_t* __restrict__ WT,
    const float*  __restrict__ bias,
    int N, int Nb2,                       // Nb2 = Ntiles/2
    bf16_t* __restrict__ outB, int outStride,
    const float* __restrict__ ew, bf16_t* __restrict__ outT,
    float* __restrict__ outF, float* __restrict__ lastF,
    bf16_t* __restrict__ gateA, int t)
{
  const int wave = threadIdx.x >> 5;
  const int lane = threadIdx.x & 31;
  const int gid  = blockIdx.x * 8 + wave;
  const int tm2  = gid / Nb2;             // 0..15  (M = 512 -> 16 blocks of 32)
  const int tn2  = gid - tm2 * Nb2;
  const int half = lane >> 4;
  const int l15  = lane & 15;
  const int m0 = tm2 * 32;
  const int n0 = tn2 * 32;

  const bf16_t* ar0 = A  + (size_t)(m0 + l15) * Ks;
  const bf16_t* ar1 = ar0 + (size_t)16 * Ks;
  const bf16_t* br0 = WT + (size_t)(n0 + l15) * Ks;
  const bf16_t* br1 = br0 + (size_t)16 * Ks;

  v8f c00 = {0.f,0.f,0.f,0.f,0.f,0.f,0.f,0.f};
  v8f c01 = c00, c10 = c00, c11 = c00;

  for (int k0 = 0; k0 < Ks; k0 += 32) {
    v16bf a0 = load_afrag(ar0, k0, half);
    v16bf a1 = load_afrag(ar1, k0, half);
    v16bf b0 = load_bfrag(br0, k0, half);
    v16bf b1 = load_bfrag(br1, k0, half);
    c00 = wmma_bf16(a0, b0, c00);
    c01 = wmma_bf16(a0, b1, c01);
    c10 = wmma_bf16(a1, b0, c10);
    c11 = wmma_bf16(a1, b1, c11);
  }

  epilogue_tile<MODE>(c00, m0,      n0,      half, l15, bias, N, outB, outStride, ew, outT, outF, lastF, gateA, t);
  epilogue_tile<MODE>(c01, m0,      n0 + 16, half, l15, bias, N, outB, outStride, ew, outT, outF, lastF, gateA, t);
  epilogue_tile<MODE>(c10, m0 + 16, n0,      half, l15, bias, N, outB, outStride, ew, outT, outF, lastF, gateA, t);
  epilogue_tile<MODE>(c11, m0 + 16, n0 + 16, half, l15, bias, N, outB, outStride, ew, outT, outF, lastF, gateA, t);
}

// gate layer 5 (256->3) + softmax + build MoE layer-1 input A1[512][96]
__global__ void gate_tail_kernel(const bf16_t* __restrict__ g4,
                                 const float* __restrict__ gW5,
                                 const float* __restrict__ gb5,
                                 const float* __restrict__ lastF,
                                 float* __restrict__ ewOut,
                                 bf16_t* __restrict__ A1)
{
  int b = blockIdx.x * blockDim.x + threadIdx.x;
  if (b >= B_ROWS) return;
  float l0 = gb5[0], l1 = gb5[1], l2 = gb5[2];
  const bf16_t* g = g4 + (size_t)b * 256;
  for (int k = 0; k < 256; ++k) {
    float gv = (float)g[k];
    l0 += gv * gW5[k * 3 + 0];
    l1 += gv * gW5[k * 3 + 1];
    l2 += gv * gW5[k * 3 + 2];
  }
  float mx = fmaxf(l0, fmaxf(l1, l2));
  float e0 = __expf(l0 - mx), e1 = __expf(l1 - mx), e2 = __expf(l2 - mx);
  float inv = 1.f / (e0 + e1 + e2);
  e0 *= inv; e1 *= inv; e2 *= inv;
  ewOut[b * 3 + 0] = e0; ewOut[b * 3 + 1] = e1; ewOut[b * 3 + 2] = e2;
  const float* lr = lastF + (size_t)b * D_IN;
  bf16_t* a = A1 + (size_t)b * 96;
  for (int km = 0; km < D_IN; ++km) {
    float x = lr[km];
    a[km]            = (bf16_t)(e0 * x);
    a[D_IN + km]     = (bf16_t)(e1 * x);
    a[2 * D_IN + km] = (bf16_t)(e2 * x);
  }
  for (int k = 81; k < 96; ++k) a[k] = (bf16_t)0.f;   // K padding
}

// t=0: copy x to output, seed lastF and gate input (bf16, K padded to 32)
__global__ void init_kernel(const float* __restrict__ x,
                            float* __restrict__ outF,
                            float* __restrict__ lastF,
                            bf16_t* __restrict__ gateA)
{
  int i = blockIdx.x * blockDim.x + threadIdx.x;   // 512*32
  if (i >= B_ROWS * 32) return;
  int b = i >> 5, k = i & 31;
  if (k < D_IN) {
    float v = x[b * D_IN + k];
    outF[(size_t)b * T_STEPS * D_IN + k] = v;
    lastF[b * D_IN + k] = v;
    gateA[i] = (bf16_t)v;
  } else {
    gateA[i] = (bf16_t)0.f;
  }
}

// W[K][N] (f32) -> WT[Npad][Ks] (bf16, transposed, zero padded)
__global__ void conv_T_kernel(const float* __restrict__ W, int K, int N,
                              int Ks, bf16_t* __restrict__ WT, int total)
{
  int i = blockIdx.x * blockDim.x + threadIdx.x;
  if (i >= total) return;
  int n = i / Ks, k = i - n * Ks;
  float v = (n < N && k < K) ? W[(size_t)k * N + n] : 0.f;
  WT[i] = (bf16_t)v;
}

// E[3][m][N] (f32) -> WT[Npad][Ks] with K = e*m + km (stacked experts)
__global__ void conv_E_kernel(const float* __restrict__ E, int m, int N,
                              int Ks, bf16_t* __restrict__ WT, int total)
{
  int i = blockIdx.x * blockDim.x + threadIdx.x;
  if (i >= total) return;
  int n = i / Ks, k = i - n * Ks;
  float v = 0.f;
  if (n < N && k < 3 * m) {
    int e = k / m, km = k - e * m;
    v = E[((size_t)e * m + km) * N + n];
  }
  WT[i] = (bf16_t)v;
}

extern "C" void kernel_launch(void* const* d_in, const int* in_sizes, int n_in,
                              void* d_out, int out_size, void* d_ws, size_t ws_size,
                              hipStream_t stream)
{
  (void)in_sizes; (void)n_in; (void)out_size; (void)ws_size;

  const float* x   = (const float*)d_in[0];
  const float* gW1 = (const float*)d_in[1];  const float* gb1 = (const float*)d_in[2];
  const float* gW2 = (const float*)d_in[3];  const float* gb2 = (const float*)d_in[4];
  const float* gW3 = (const float*)d_in[5];  const float* gb3 = (const float*)d_in[6];
  const float* gW4 = (const float*)d_in[7];  const float* gb4 = (const float*)d_in[8];
  const float* gW5 = (const float*)d_in[9];  const float* gb5 = (const float*)d_in[10];
  const float* E1  = (const float*)d_in[11]; const float* b1  = (const float*)d_in[12];
  const float* E2  = (const float*)d_in[13]; const float* b2  = (const float*)d_in[14];
  const float* E3  = (const float*)d_in[15]; const float* b3  = (const float*)d_in[16];
  const float* E4  = (const float*)d_in[17]; const float* b4  = (const float*)d_in[18];
  const float* E5  = (const float*)d_in[19]; const float* b5  = (const float*)d_in[20];
  float* outF = (float*)d_out;

  // ---- workspace layout (256B aligned slices) ----
  char* p = (char*)d_ws;
  auto alloc = [&](size_t bytes) -> void* {
    void* r = (void*)p; p += (bytes + 255) & ~(size_t)255; return r;
  };
  bf16_t* WTg1 = (bf16_t*)alloc((size_t)1024 * 32  * 2);
  bf16_t* WTg2 = (bf16_t*)alloc((size_t)512 * 1024 * 2);
  bf16_t* WTg3 = (bf16_t*)alloc((size_t)256 * 512  * 2);
  bf16_t* WTg4 = (bf16_t*)alloc((size_t)256 * 256  * 2);
  bf16_t* WTm1 = (bf16_t*)alloc((size_t)1024 * 96  * 2);
  bf16_t* WTm2 = (bf16_t*)alloc((size_t)512 * 3072 * 2);
  bf16_t* WTm3 = (bf16_t*)alloc((size_t)512 * 1536 * 2);
  bf16_t* WTm4 = (bf16_t*)alloc((size_t)256 * 1536 * 2);
  bf16_t* WTm5 = (bf16_t*)alloc((size_t)32  * 768  * 2);
  bf16_t* gateA = (bf16_t*)alloc((size_t)B_ROWS * 32   * 2);
  bf16_t* g1    = (bf16_t*)alloc((size_t)B_ROWS * 1024 * 2);
  bf16_t* g2    = (bf16_t*)alloc((size_t)B_ROWS * 512  * 2);
  bf16_t* g3    = (bf16_t*)alloc((size_t)B_ROWS * 256  * 2);
  bf16_t* g4    = (bf16_t*)alloc((size_t)B_ROWS * 256  * 2);
  float*  ew    = (float*)alloc((size_t)B_ROWS * 3 * 4);
  bf16_t* A1    = (bf16_t*)alloc((size_t)B_ROWS * 96   * 2);
  bf16_t* A2    = (bf16_t*)alloc((size_t)B_ROWS * 3072 * 2);
  bf16_t* A3    = (bf16_t*)alloc((size_t)B_ROWS * 1536 * 2);
  bf16_t* A4    = (bf16_t*)alloc((size_t)B_ROWS * 1536 * 2);
  bf16_t* A5    = (bf16_t*)alloc((size_t)B_ROWS * 768  * 2);
  float*  lastF = (float*)alloc((size_t)B_ROWS * D_IN * 4);

  // ---- per-launch weight conversion: bf16 + transpose + pad ----
  auto convT = [&](const float* W, int K, int N, int Ks, int Npad, bf16_t* WT) {
    int total = Npad * Ks;
    conv_T_kernel<<<(total + 255) / 256, 256, 0, stream>>>(W, K, N, Ks, WT, total);
  };
  auto convE = [&](const float* E, int m, int N, int Ks, int Npad, bf16_t* WT) {
    int total = Npad * Ks;
    conv_E_kernel<<<(total + 255) / 256, 256, 0, stream>>>(E, m, N, Ks, WT, total);
  };
  convT(gW1, 27,   1024, 32,   1024, WTg1);
  convT(gW2, 1024, 512,  1024, 512,  WTg2);
  convT(gW3, 512,  256,  512,  256,  WTg3);
  convT(gW4, 256,  256,  256,  256,  WTg4);
  convE(E1, 27,   1024, 96,   1024, WTm1);
  convE(E2, 1024, 512,  3072, 512,  WTm2);
  convE(E3, 512,  512,  1536, 512,  WTm3);
  convE(E4, 512,  256,  1536, 256,  WTm4);
  convE(E5, 256,  27,   768,  32,   WTm5);

  init_kernel<<<(B_ROWS * 32 + 255) / 256, 256, 0, stream>>>(x, outF, lastF, gateA);

  // grid: waves = 16 * (Ntiles/2), 8 waves per block -> blocks = Ntiles
  // ---- 84 autoregressive steps; stream order = layer dependency ----
  for (int t = 1; t < T_STEPS; ++t) {
    // gating MLP
    gemm_bf16_wmma<0><<<64, 256, 0, stream>>>(gateA, 32, WTg1, gb1, 1024, 32,
        g1, 1024, nullptr, nullptr, nullptr, lastF, gateA, t);
    gemm_bf16_wmma<0><<<32, 256, 0, stream>>>(g1, 1024, WTg2, gb2, 512, 16,
        g2, 512, nullptr, nullptr, nullptr, lastF, gateA, t);
    gemm_bf16_wmma<0><<<16, 256, 0, stream>>>(g2, 512, WTg3, gb3, 256, 8,
        g3, 256, nullptr, nullptr, nullptr, lastF, gateA, t);
    gemm_bf16_wmma<1><<<16, 256, 0, stream>>>(g3, 256, WTg4, gb4, 256, 8,
        g4, 256, nullptr, nullptr, nullptr, lastF, gateA, t);
    gate_tail_kernel<<<2, 256, 0, stream>>>(g4, gW5, gb5, lastF, ew, A1);
    // MoE layers: one GEMM each (experts stacked in K, ew folded into A)
    gemm_bf16_wmma<2><<<64, 256, 0, stream>>>(A1, 96, WTm1, b1, 1024, 32,
        nullptr, 0, ew, A2, nullptr, lastF, gateA, t);
    gemm_bf16_wmma<2><<<32, 256, 0, stream>>>(A2, 3072, WTm2, b2, 512, 16,
        nullptr, 0, ew, A3, nullptr, lastF, gateA, t);
    gemm_bf16_wmma<2><<<32, 256, 0, stream>>>(A3, 1536, WTm3, b3, 512, 16,
        nullptr, 0, ew, A4, nullptr, lastF, gateA, t);
    gemm_bf16_wmma<2><<<16, 256, 0, stream>>>(A4, 1536, WTm4, b4, 256, 8,
        nullptr, 0, ew, A5, nullptr, lastF, gateA, t);
    gemm_bf16_wmma<3><<<2, 256, 0, stream>>>(A5, 768, WTm5, b5, 27, 1,
        nullptr, 0, nullptr, nullptr, outF, lastF, gateA, t);
  }
}
